// RNN_14345190768814
// MI455X (gfx1250) — matrix operand order repaired and checked by reading, not compiled, not run on hardware
//
#include <hip/hip_runtime.h>
#include <hip/hip_bf16.h>
#include <math.h>
#include <stdint.h>

// Problem constants (from reference): B=256, T=512, D=512, H=1024
#define RNN_B 256
#define RNN_T 512
#define RNN_D 512
#define RNN_H 1024
#define RNN_KTOT (RNN_D + RNN_H)   // 1536

// LDS pitches (elements).  All chosen so row stride in dwords % 64 == 4,
// giving the conflict-minimal 32-lane ds_load_b128 pattern.
#define A_PITCH  (RNN_KTOT + 8)    // fused path: 772 dwords
#define X_PITCH  (RNN_D + 8)       // xp-gemm:    260 dwords
#define H_PITCH  (RNN_H + 8)       // xp-step:    516 dwords

typedef __bf16 bf16_t;
typedef __attribute__((ext_vector_type(16))) __bf16 v16bf;
typedef __attribute__((ext_vector_type(8)))  __bf16 v8bf;
typedef __attribute__((ext_vector_type(8)))  float  v8f;

// ---------------------------------------------------------------------------
// Kernel 1: transpose + convert W_i2h (KTOT x H, f32 row-major) into
// WT (H x KTOT, bf16 row-major) so a WMMA B-fragment (16 consecutive K for
// one output column) is one contiguous 32-byte load.  Zeroes h0 (f32 + bf16).
// ---------------------------------------------------------------------------
__global__ void rnn_prep_kernel(const float* __restrict__ W,
                                bf16_t* __restrict__ WT,
                                float* __restrict__ h0,
                                bf16_t* __restrict__ h0bf) {
    int i = blockIdx.x * blockDim.x + threadIdx.x;
    if (i < RNN_H * RNN_KTOT) {
        int n = i / RNN_KTOT;      // output column (0..H-1)
        int k = i % RNN_KTOT;      // input row    (0..KTOT-1)
        WT[i] = (bf16_t)W[(size_t)k * RNN_H + n];
    }
    if (i < RNN_B * RNN_H) {
        h0[i]   = 0.0f;
        h0bf[i] = (bf16_t)0.0f;
    }
}

// Shared WMMA helper: one 16x16 tile, K-loop over [0, klen) reading A rows
// from LDS and B rows from wrow (contiguous bf16).
__device__ __forceinline__ v8f wmma_tile(const bf16_t* arow,
                                         const bf16_t* wrow, int klen, int hi) {
    v8f acc = {};
#pragma unroll 4
    for (int k0 = 0; k0 < klen; k0 += 32) {
        const int ka = k0 + hi * 16;
        v8bf alo = *(const v8bf*)(arow + ka);
        v8bf ahi = *(const v8bf*)(arow + ka + 8);
        v16bf a = __builtin_shufflevector(alo, ahi,
                                          0, 1, 2, 3, 4, 5, 6, 7,
                                          8, 9, 10, 11, 12, 13, 14, 15);
        v16bf b = *(const v16bf*)(wrow + k0 + hi * 16);
        acc = __builtin_amdgcn_wmma_f32_16x16x32_bf16(
            false, a, false, b, (short)0, acc, false, false);
    }
    return acc;
}

// ---------------------------------------------------------------------------
// Kernel 2a (big-ws path): XP = xs @ Wx + b_i2h, stored bf16 as (T*B, H).
// Output row r = t*B + b.  Grid (T*B/16, H/128), 8 waves; fully parallel.
// ---------------------------------------------------------------------------
__global__ __launch_bounds__(256)
void rnn_xp_gemm_kernel(const float* __restrict__ x,      // (B,T,D) f32
                        const bf16_t* __restrict__ WT,    // (H,KTOT) bf16
                        const float* __restrict__ bias,   // (H,) f32
                        bf16_t* __restrict__ XP) {        // (T*B,H) bf16
    __shared__ __align__(16) bf16_t As[16 * X_PITCH];     // ~16.6 KB

    const int tid     = threadIdx.x;
    const int wave    = tid >> 5;
    const int lane    = tid & 31;
    const int rowlane = lane & 15;
    const int hi      = lane >> 4;

    const int r0 = blockIdx.x * 16;      // output row strip (same t for all 16)
    const int t  = r0 >> 8;              // r0 / B
    const int b0 = r0 & (RNN_B - 1);     // r0 % B

    // Stage x[b0..b0+15][t][0..511] -> bf16 LDS (1024 groups of 8; 4/thread)
    for (int g = tid; g < 16 * (RNN_D / 8); g += 256) {
        const int row = g / (RNN_D / 8);
        const int k   = (g % (RNN_D / 8)) * 8;
        const float* src = x + ((size_t)(b0 + row) * RNN_T + (size_t)t) * RNN_D + k;
        v8f f = *(const v8f*)src;
        v8bf o;
#pragma unroll
        for (int i = 0; i < 8; ++i) o[i] = (bf16_t)f[i];
        *(v8bf*)&As[row * X_PITCH + k] = o;
    }
    __syncthreads();

    const int n0   = blockIdx.y * 128 + wave * 16;
    const int bcol = n0 + rowlane;
    const bf16_t* arow = &As[rowlane * X_PITCH];
    const bf16_t* wrow = WT + (size_t)bcol * RNN_KTOT;    // Wx part: k in [0,512)

    v8f acc = wmma_tile(arow, wrow, RNN_D, hi);

    const float bv = bias[bcol];
#pragma unroll
    for (int r = 0; r < 8; ++r) {
        const int lr = hi * 8 + r;
        XP[(size_t)(r0 + lr) * RNN_H + bcol] = (bf16_t)(acc[r] + bv);
    }
}

// ---------------------------------------------------------------------------
// Kernel 2b (big-ws path): h_new = XP[t] + h_prev @ Wh.  K = 1024 only.
// A-strip staging is a raw bf16 copy done with GLOBAL_LOAD_ASYNC_TO_LDS_B128
// (ASYNCcnt), since h is kept in bf16 between steps anyway (numerically
// identical to convert-on-read).  f32 h is also written for the final output.
// ---------------------------------------------------------------------------
__global__ __launch_bounds__(256)
void rnn_step_xp_kernel(const bf16_t* __restrict__ XP,      // (T*B,H) bf16
                        const bf16_t* __restrict__ WT,      // (H,KTOT) bf16
                        const bf16_t* __restrict__ hprevbf, // (B,H) bf16
                        float* __restrict__ hnext,          // (B,H) f32
                        bf16_t* __restrict__ hnextbf,       // (B,H) bf16
                        int t) {
    __shared__ __align__(16) bf16_t Hs[16 * H_PITCH];       // ~33 KB

    const int tid     = threadIdx.x;
    const int wave    = tid >> 5;
    const int lane    = tid & 31;
    const int rowlane = lane & 15;
    const int hi      = lane >> 4;

    const int m0 = blockIdx.x * 16;

    // ---- Async stage: h_prev(bf16) rows m0..m0+15 -> LDS, 16B granules ----
    // 16 rows x 128 granules = 2048; 8 per thread.  Row pad preserved by
    // computing the LDS address per granule.
    for (int g = tid; g < 16 * (RNN_H / 8); g += 256) {
        const int row = g >> 7;            // granule / 128
        const int ke  = (g & 127) * 8;     // element offset within row
        uint64_t ga = (uint64_t)(uintptr_t)(hprevbf +
                        (size_t)(m0 + row) * RNN_H + ke);
        uint32_t la = (uint32_t)(uintptr_t)(&Hs[row * H_PITCH + ke]);
        asm volatile("global_load_async_to_lds_b128 %0, %1, off"
                     :: "v"(la), "v"(ga)
                     : "memory");
    }
    asm volatile("s_wait_asynccnt 0x0" ::: "memory");
    __syncthreads();

    const int n0   = blockIdx.y * 128 + wave * 16;
    const int bcol = n0 + rowlane;
    const bf16_t* arow = &Hs[rowlane * H_PITCH];
    const bf16_t* wrow = WT + (size_t)bcol * RNN_KTOT + RNN_D;  // Wh part

    v8f acc = wmma_tile(arow, wrow, RNN_H, hi);

    // Epilogue: add precomputed XP[t] tile; write f32 and bf16 h.
    const bf16_t* xprow = XP + ((size_t)t * RNN_B + m0) * RNN_H + bcol;
#pragma unroll
    for (int r = 0; r < 8; ++r) {
        const int lr = hi * 8 + r;
        float val = acc[r] + (float)xprow[(size_t)lr * RNN_H];
        hnext[(size_t)(m0 + lr) * RNN_H + bcol]   = val;
        hnextbf[(size_t)(m0 + lr) * RNN_H + bcol] = (bf16_t)val;
    }
}

// ---------------------------------------------------------------------------
// Kernel 2c (fallback, small ws): fused h_new = [x_t | h_prev] @ W_i2h + b.
// ---------------------------------------------------------------------------
__global__ __launch_bounds__(256)
void rnn_step_fused_kernel(const float* __restrict__ x,
                           const bf16_t* __restrict__ WT,
                           const float* __restrict__ bias,
                           const float* __restrict__ hprev,
                           float* __restrict__ hnext,
                           int t) {
    __shared__ __align__(16) bf16_t Atile[16 * A_PITCH];  // ~48.3 KB

    const int tid     = threadIdx.x;
    const int wave    = tid >> 5;
    const int lane    = tid & 31;
    const int rowlane = lane & 15;
    const int hi      = lane >> 4;

    const int m0 = blockIdx.x * 16;

    for (int g = tid; g < 16 * (RNN_KTOT / 8); g += 256) {
        const int row = g / (RNN_KTOT / 8);
        const int k   = (g % (RNN_KTOT / 8)) * 8;
        const int br  = m0 + row;
        const float* src;
        if (k < RNN_D) {
            src = x + ((size_t)br * RNN_T + (size_t)t) * RNN_D + k;
        } else {
            src = hprev + (size_t)br * RNN_H + (k - RNN_D);
        }
        v8f f = *(const v8f*)src;
        v8bf o;
#pragma unroll
        for (int i = 0; i < 8; ++i) o[i] = (bf16_t)f[i];
        *(v8bf*)&Atile[row * A_PITCH + k] = o;
    }
    __syncthreads();

    const int n0   = blockIdx.y * 128 + wave * 16;
    const int bcol = n0 + rowlane;
    const bf16_t* arow = &Atile[rowlane * A_PITCH];
    const bf16_t* wrow = WT + (size_t)bcol * RNN_KTOT;

    v8f acc = wmma_tile(arow, wrow, RNN_KTOT, hi);

    const float bv = bias[bcol];
#pragma unroll
    for (int r = 0; r < 8; ++r) {
        const int row = m0 + hi * 8 + r;
        hnext[(size_t)row * RNN_H + bcol] = acc[r] + bv;
    }
}

// ---------------------------------------------------------------------------
// Kernel 3: out[b] = sigmoid(h[b,:] . W_h2o + b_h2o); copy hidden to d_out.
// ---------------------------------------------------------------------------
__global__ __launch_bounds__(256)
void rnn_final_kernel(const float* __restrict__ h,
                      const float* __restrict__ W_h2o,
                      const float* __restrict__ b_h2o,
                      float* __restrict__ out) {
    const int b   = blockIdx.x;
    const int tid = threadIdx.x;
    __shared__ float red[256];

    float s = 0.0f;
    for (int i = tid; i < RNN_H; i += 256) {
        float v = h[(size_t)b * RNN_H + i];
        s += v * W_h2o[i];
        out[RNN_B + (size_t)b * RNN_H + i] = v;
    }
    red[tid] = s;
    __syncthreads();
    for (int off = 128; off > 0; off >>= 1) {
        if (tid < off) red[tid] += red[tid + off];
        __syncthreads();
    }
    if (tid == 0) {
        float z = red[0] + b_h2o[0];
        out[b] = 1.0f / (1.0f + __expf(-z));
    }
}

// ---------------------------------------------------------------------------
extern "C" void kernel_launch(void* const* d_in, const int* in_sizes, int n_in,
                              void* d_out, int out_size, void* d_ws, size_t ws_size,
                              hipStream_t stream) {
    const float* x     = (const float*)d_in[0];   // (B,T,D)
    const float* W_i2h = (const float*)d_in[1];   // (KTOT,H)
    const float* b_i2h = (const float*)d_in[2];   // (H,)
    const float* W_h2o = (const float*)d_in[3];   // (H,1)
    const float* b_h2o = (const float*)d_in[4];   // (1,)
    float* out = (float*)d_out;

    const size_t WT_BYTES  = (size_t)RNN_H * RNN_KTOT * 2;         // 3 MB
    const size_t H_BYTES   = (size_t)RNN_B * RNN_H * 4;            // 1 MB
    const size_t HBF_BYTES = (size_t)RNN_B * RNN_H * 2;            // 0.5 MB
    const size_t XP_BYTES  = (size_t)RNN_T * RNN_B * RNN_H * 2;    // 256 MB

    char* ws = (char*)d_ws;
    bf16_t* WT   = (bf16_t*)ws;
    float*  hA   = (float*) (ws + WT_BYTES);
    float*  hB   = (float*) (ws + WT_BYTES + H_BYTES);
    bf16_t* hbfA = (bf16_t*)(ws + WT_BYTES + 2 * H_BYTES);
    bf16_t* hbfB = (bf16_t*)(ws + WT_BYTES + 2 * H_BYTES + HBF_BYTES);
    bf16_t* XP   = (bf16_t*)(ws + WT_BYTES + 2 * H_BYTES + 2 * HBF_BYTES);

    // 1) Weight transpose/convert + zero h0 (f32 and bf16)
    {
        int total  = RNN_H * RNN_KTOT;
        int blocks = (total + 255) / 256;
        rnn_prep_kernel<<<blocks, 256, 0, stream>>>(W_i2h, WT, hA, hbfA);
    }

    float*  curF = hA;   float*  nxtF = hB;
    bf16_t* curB = hbfA; bf16_t* nxtB = hbfB;
    dim3 gridS(RNN_B / 16, RNN_H / 128);            // (16,8), 8 waves/block

    if (ws_size >= WT_BYTES + 2 * H_BYTES + 2 * HBF_BYTES + XP_BYTES) {
        // 2a) Hoist the parallel GEMM: XP = xs@Wx + b  (full-device WMMA)
        dim3 gridG((RNN_T * RNN_B) / 16, RNN_H / 128);  // (8192,8)
        rnn_xp_gemm_kernel<<<gridG, 256, 0, stream>>>(x, WT, b_i2h, XP);
        // 2b) 512 dependent steps, K = 1024, async-to-LDS A staging
        for (int t = 0; t < RNN_T; ++t) {
            rnn_step_xp_kernel<<<gridS, 256, 0, stream>>>(XP, WT, curB,
                                                          nxtF, nxtB, t);
            float*  tf = curF; curF = nxtF; nxtF = tf;
            bf16_t* tb = curB; curB = nxtB; nxtB = tb;
        }
    } else {
        // 2c) Fallback: fused steps, K = 1536, f32 h ping-pong
        for (int t = 0; t < RNN_T; ++t) {
            rnn_step_fused_kernel<<<gridS, 256, 0, stream>>>(x, WT, b_i2h,
                                                             curF, nxtF, t);
            float* tmp = curF; curF = nxtF; nxtF = tmp;
        }
    }

    // 3) Output head + hidden copy
    rnn_final_kernel<<<RNN_B, 256, 0, stream>>>(curF, W_h2o, b_h2o, out);
}
